// DualRealConvTasNet_16561393893442
// MI455X (gfx1250) — matrix operand order repaired and checked by reading, not compiled, not run on hardware
//
#include <hip/hip_runtime.h>

typedef _Float16 h16;
typedef __attribute__((ext_vector_type(16))) _Float16 v16h;
typedef __attribute__((ext_vector_type(8)))  _Float16 v8h;
typedef __attribute__((ext_vector_type(8)))  float    v8f;
typedef __attribute__((ext_vector_type(4)))  unsigned int v4u;
typedef __attribute__((ext_vector_type(4)))  int v4i;
typedef __attribute__((ext_vector_type(8)))  int v8i;

#define T_RAW 16000
#define T0 1999   // after enc conv0 (stride 8, k16)
#define T1 1997   // after enc conv1 (k3)
#define T2 1995   // after enc conv2 (k3) == TCN length
#define NBLK 24
#define MAX_LDS_H 9472   // halfs: covers (16*NT+taps-1)*(Cin+8) worst case 18*520=9360

#if defined(__HIP_DEVICE_COMPILE__) && __has_builtin(__builtin_amdgcn_tensor_load_to_lds) && __has_builtin(__builtin_amdgcn_s_wait_tensorcnt)
#define USE_TDM 1
#else
#define USE_TDM 0
#endif

// ---------------------------------------------------------------------------
__device__ __forceinline__ float act_apply(float x, int act, float a) {
  if (act == 1) return x >= 0.f ? x : a * x;
  if (act == 2) return 1.f / (1.f + __expf(-x));
  return x;
}

__device__ __forceinline__ v16h ld_frag16(const h16* p) {  // 16B-aligned pair load
  v8h lo = *(const v8h*)p;
  v8h hi = *(const v8h*)(p + 16);
  v16h r;
#pragma unroll
  for (int j = 0; j < 8; ++j) { r[j] = lo[j]; r[j + 8] = hi[j]; }
  return r;
}

// ---------------------------------------------------------------------------
// Conv-as-GEMM, WMMA f16 / fp32 accum, LDS-staged activations.
// Activations time-major f16: B[b][t][Cin]; weights pre-packed A fragments.
// Block = 8 waves covering MT m-tiles x NT t-subtiles; LDS window staged by
// TDM (tensor_load_to_lds) when available, else cooperative loads.
__global__ __launch_bounds__(256)
void gemm_conv_wmma(const h16* __restrict__ Bact, const h16* __restrict__ Apack,
                    const float* __restrict__ bias, const float* __restrict__ accsrc,
                    float* __restrict__ out32, h16* __restrict__ out16,
                    const float* __restrict__ alpha32, const float* __restrict__ alpha16,
                    int BS, int Cin, int M, int Tin, int Tout, int taps, int pad,
                    int act32, int act16, int MT, int NT)
{
  __shared__ h16 sh[MAX_LDS_H];
  const int tid  = threadIdx.x;
  const int wave = tid >> 5, lane = tid & 31;
  const int n = lane & 15, hh = lane >> 4;
  const int b = blockIdx.z;
  const int ldsrow = Cin + 8;                 // +16B pad per row (bank stagger)
  const int R = 16 * NT + taps - 1;           // staged time rows
  const int tbase = blockIdx.x * NT * 16;
  const int cshift = 31 - __clz(Cin);

  // ---- stage activation window [tbase-pad, tbase-pad+R) into LDS ----
#if USE_TDM
  {
    int lead = pad - tbase; if (lead < 0) lead = 0;       // rows with t<0 -> zero
    for (int v = tid; (v << 3) < (lead << cshift); v += 256) {
      int idx = v << 3;
      int r = idx >> cshift, c = idx & (Cin - 1);
      v8h z;
#pragma unroll
      for (int j = 0; j < 8; ++j) z[j] = (h16)0.f;
      *(v8h*)(sh + r * ldsrow + c) = z;
    }
    if (wave == 0) {
      const int tstart = tbase - pad + lead;
      const unsigned long long gaddr =
          (unsigned long long)(uintptr_t)(Bact + ((size_t)b * Tin + tstart) * Cin);
      const unsigned lds_addr =
          (unsigned)(uintptr_t)sh + (unsigned)(lead * ldsrow * 2);
      const int rows  = R - lead;                 // tile_dim1
      const int vdim1 = Tin - tstart;             // valid rows; beyond -> zero-fill
      const unsigned padi = (unsigned)(cshift - 2); // pad every Cin/2 DWORDs
      v4u g0; v8i g1; v4i g2 = {}; v4i g3 = {};
      g0[0] = 1u;                                  // count=1, user mode
      g0[1] = lds_addr;
      g0[2] = (unsigned)gaddr;
      g0[3] = (unsigned)((gaddr >> 32) & 0x1FFFFFFull) | (2u << 30);  // type=2
      g1[0] = (int)((1u << 16) | (1u << 20) | (padi << 22) | (3u << 25));
      //        data_size=2B    pad_enable    pad_interval    pad_amount=4 DW
      g1[1] = (int)(((unsigned)Cin & 0xFFFFu) << 16);                     // dim0 lo
      g1[2] = (int)(((unsigned)Cin >> 16) | (((unsigned)vdim1 & 0xFFFFu) << 16));
      g1[3] = (int)(((unsigned)vdim1 >> 16) | ((unsigned)Cin << 16));     // tile_dim0
      g1[4] = (int)((unsigned)rows & 0xFFFFu);                            // tile_dim1
      g1[5] = (int)Cin;                                                   // dim0 stride
      g1[6] = 0;
      g1[7] = 0;
#if __clang_major__ >= 23
      v8i g4 = {};
      __builtin_amdgcn_tensor_load_to_lds(g0, g1, g2, g3, g4, 0);
#else
      __builtin_amdgcn_tensor_load_to_lds(g0, g1, g2, g3, 0);
#endif
      __builtin_amdgcn_s_wait_tensorcnt(0);
    }
  }
#else
  for (int v = tid; (v << 3) < (R << cshift); v += 256) {
    int idx = v << 3;
    int r = idx >> cshift, c = idx & (Cin - 1);
    int tg = tbase + r - pad;
    v8h val;
    if (tg >= 0 && tg < Tin) {
      val = *(const v8h*)(Bact + ((size_t)b * Tin + tg) * Cin + c);
    } else {
#pragma unroll
      for (int j = 0; j < 8; ++j) val[j] = (h16)0.f;
    }
    *(v8h*)(sh + r * ldsrow + c) = val;
  }
#endif
  __syncthreads();

  // ---- per-wave 16x16 tile ----
  const int mtile = blockIdx.y * MT + (wave % MT);
  const int nsub  = wave / MT;
  const int t_out = tbase + nsub * 16 + n;
  const int KC    = Cin >> 5;
  const int mrow  = mtile * 16 + hh * 8;

  v8f acc = {};
  if (bias) {
#pragma unroll
    for (int j = 0; j < 8; ++j) acc[j] = bias[mrow + j];
  }
  if (accsrc) {
    if (t_out < Tout) {
      const float* p = accsrc + ((size_t)b * Tout + t_out) * M + mrow;
#pragma unroll
      for (int j = 0; j < 8; ++j) acc[j] += p[j];
    }
  }

  for (int tap = 0; tap < taps; ++tap) {
    const h16* arow = Apack + ((((size_t)mtile * taps + tap) * KC) * 32 + lane) * 16;
    const h16* brow = sh + (nsub * 16 + n + tap) * ldsrow + hh * 8;
    for (int kc = 0; kc < KC; ++kc) {
      v16h afrag = *(const v16h*)(arow + (size_t)kc * 512);
      if (kc + 1 < KC) __builtin_prefetch(arow + (size_t)(kc + 1) * 512, 0, 0);
      v16h bfrag = ld_frag16(brow + kc * 32);
      acc = __builtin_amdgcn_wmma_f32_16x16x32_f16(false, afrag, false, bfrag,
                                                   (short)0, acc, false, false);
    }
  }

  if (t_out < Tout) {
    const float a32 = alpha32 ? *alpha32 : 0.f;
    const float a16 = alpha16 ? *alpha16 : 0.f;
    if (out32) {
      float* p = out32 + ((size_t)b * Tout + t_out) * M + mrow;
#pragma unroll
      for (int j = 0; j < 8; ++j) p[j] = act_apply(acc[j], act32, a32);
    }
    if (out16) {
      v8h hv;
#pragma unroll
      for (int j = 0; j < 8; ++j) hv[j] = (h16)act_apply(acc[j], act16, a16);
      *(v8h*)(out16 + ((size_t)b * Tout + t_out) * M + mrow) = hv;
    }
  }
}

// ---------------------------------------------------------------------------
// Pack fp32 weights into per-lane WMMA A-fragment layout (f16).
// dst id = (((mtile*taps+tap)*KC + kc)*32 + lane)*16 + j
__global__ void pack_frag_kernel(const float* __restrict__ src0,
                                 const float* __restrict__ src1,
                                 h16* __restrict__ dst,
                                 int M, int K, int taps,
                                 long long sm, long long sk, long long st,
                                 long long base, long long total)
{
  long long id = (long long)blockIdx.x * 256 + threadIdx.x;
  if (id >= total) return;
  int j    = (int)(id & 15);
  int lane = (int)((id >> 4) & 31);
  long long rest = id >> 9;
  int KC = K >> 5;
  int kc  = (int)(rest % KC); rest /= KC;
  int tap = (int)(rest % taps);
  int mtile = (int)(rest / taps);
  int m  = mtile * 16 + (lane & 15);
  int hh = lane >> 4;
  int kk = (j < 8) ? (hh * 8 + j) : (16 + hh * 8 + (j - 8));
  int k  = kc * 32 + kk;
  float v = 0.f;
  if (m < M && k < K) {
    long long off = base + (long long)m * sm + (long long)k * sk + (long long)tap * st;
    v = src0[off];
    if (src1) v += src1[off];
  }
  dst[id] = (h16)v;
}

__global__ void pack_w3_kernel(const float* __restrict__ w3, h16* __restrict__ dst) {
  int id = blockIdx.x * 256 + threadIdx.x;
  if (id >= 16 * 512) return;
  int c = id & 511, kk = id >> 9;
  dst[id] = (h16)w3[c * 16 + kk];
}

__global__ void comb_bias_kernel(const float* __restrict__ rb, const float* __restrict__ sb,
                                 float* __restrict__ cb) {
  int id = blockIdx.x * 256 + threadIdx.x;
  if (id >= NBLK * 128) return;
  float v = sb[id];
  if ((id >> 7) < NBLK - 1) v += rb[id];
  cb[id] = v;
}

// ---------------------------------------------------------------------------
__global__ void enc0_kernel(const float* __restrict__ xr, const float* __restrict__ xi,
                            const float* __restrict__ w0, h16* __restrict__ out) {
  size_t id = (size_t)blockIdx.x * 256 + threadIdx.x;
  if (id >= (size_t)8 * T0 * 512) return;
  int o = (int)(id & 511);
  int t = (int)((id >> 9) % T0);
  int bb = (int)(id / ((size_t)T0 * 512));
  const float* x = (bb < 4) ? (xr + (size_t)bb * T_RAW) : (xi + (size_t)(bb - 4) * T_RAW);
  const float* w = w0 + o * 16;
  const float* xp = x + t * 8;
  float s = 0.f;
#pragma unroll
  for (int k = 0; k < 16; ++k) s += w[k] * xp[k];
  out[id] = (h16)s;
}

__global__ void depthwise_kernel(const h16* __restrict__ in, const float* __restrict__ dw,
                                 const float* __restrict__ db, const float* __restrict__ alpha,
                                 h16* __restrict__ out, int T, int dil) {
  size_t id = (size_t)blockIdx.x * 256 + threadIdx.x;
  if (id >= (size_t)8 * T * 32) return;
  int c = (int)(id & 31);
  size_t bt = id >> 5;
  int t = (int)(bt % T);
  int b = (int)(bt / T);
  const float a = *alpha;
  float s = db[c];
  const h16* base = in + ((size_t)b * T) * 32 + c;
#pragma unroll
  for (int k = 0; k < 3; ++k) {
    int tt = t + (k - 1) * dil;
    if (tt >= 0 && tt < T) s += dw[c * 3 + k] * (float)base[(size_t)tt * 32];
  }
  out[id] = (h16)(s >= 0.f ? s : a * s);
}

__global__ void mask_kernel(const h16* __restrict__ m16, const h16* __restrict__ F,
                            h16* __restrict__ Y) {
  size_t id = (size_t)blockIdx.x * 256 + threadIdx.x;
  if (id >= (size_t)16 * T2 * 512) return;
  int c = (int)(id & 511);
  size_t st = id >> 9;
  int t = (int)(st % T2);
  int s = (int)(st / T2);
  int b = s >> 1, src = s & 1;
  float mv = (float)m16[((size_t)b * T2 + t) * 1024 + src * 512 + c];
  float fv = (float)F[((size_t)b * T2 + t) * 512 + c];
  Y[id] = (h16)(mv * fv);
}

__global__ __launch_bounds__(256)
void dec_final_kernel(const h16* __restrict__ Y2, const h16* __restrict__ w3p,
                      float* __restrict__ out) {
  int wid  = blockIdx.x * 8 + (threadIdx.x >> 5);
  int lane = threadIdx.x & 31;
  int t = wid % T_RAW;
  int s = wid / T_RAW;
  float sum = 0.f;
#pragma unroll
  for (int tapi = 0; tapi < 2; ++tapi) {
    int u = (t >> 3) - tapi;
    int kk = t - 8 * u;
    if (u >= 0 && u < T0 && kk < 16) {
      const h16* xrow = Y2 + ((size_t)s * T0 + u) * 512 + lane * 16;
      const h16* wrow = w3p + kk * 512 + lane * 16;
#pragma unroll
      for (int j = 0; j < 16; ++j) sum += (float)xrow[j] * (float)wrow[j];
    }
  }
#pragma unroll
  for (int off = 16; off > 0; off >>= 1) sum += __shfl_down(sum, off, 32);
  if (lane == 0) out[(size_t)s * T_RAW + t] = sum;
}

// ---------------------------------------------------------------------------
static inline void launch_gemm(hipStream_t s, const h16* B, const h16* A,
                               const float* bias, const float* accsrc,
                               float* out32, h16* out16,
                               const float* a32, const float* a16,
                               int BS, int Cin, int M, int Tin, int Tout,
                               int taps, int pad, int act32, int act16) {
  int MT = (M / 16 < 8) ? (M / 16) : 8;
  int NT = 8 / MT;
  dim3 grid((Tout + 16 * NT - 1) / (16 * NT), M / (16 * MT), BS);
  gemm_conv_wmma<<<grid, 256, 0, s>>>(B, A, bias, accsrc, out32, out16, a32, a16,
                                      BS, Cin, M, Tin, Tout, taps, pad, act32, act16,
                                      MT, NT);
}

static inline void launch_pack(hipStream_t s, const float* s0, const float* s1, h16* dst,
                               int M, int K, int taps,
                               long long sm, long long sk, long long st, long long base) {
  long long total = (long long)M * K * taps;
  pack_frag_kernel<<<(unsigned)((total + 255) / 256), 256, 0, s>>>(s0, s1, dst, M, K, taps,
                                                                   sm, sk, st, base, total);
}

extern "C" void kernel_launch(void* const* d_in, const int* in_sizes, int n_in,
                              void* d_out, int out_size, void* d_ws, size_t ws_size,
                              hipStream_t stream) {
  (void)in_sizes; (void)n_in; (void)out_size; (void)ws_size;
  const float* x_real  = (const float*)d_in[0];
  const float* x_imag  = (const float*)d_in[1];
  const float* enc_w0  = (const float*)d_in[2];
  const float* enc_w1  = (const float*)d_in[3];
  const float* enc_w2  = (const float*)d_in[4];
  const float* enc_a   = (const float*)d_in[5];
  const float* mg_in_w = (const float*)d_in[6];
  const float* mg_in_b = (const float*)d_in[7];
  const float* blk_w1  = (const float*)d_in[8];
  const float* blk_b1  = (const float*)d_in[9];
  const float* blk_a1  = (const float*)d_in[10];
  const float* blk_dw  = (const float*)d_in[11];
  const float* blk_db  = (const float*)d_in[12];
  const float* blk_a2  = (const float*)d_in[13];
  const float* blk_rw  = (const float*)d_in[14];
  const float* blk_rb  = (const float*)d_in[15];
  const float* blk_sw  = (const float*)d_in[16];
  const float* blk_sb  = (const float*)d_in[17];
  const float* mg_out_a = (const float*)d_in[18];
  const float* mg_out_w = (const float*)d_in[19];
  const float* mg_out_b = (const float*)d_in[20];
  const float* dec_w1  = (const float*)d_in[21];
  const float* dec_w2  = (const float*)d_in[22];
  const float* dec_a   = (const float*)d_in[23];
  const float* dec_w3  = (const float*)d_in[24];
  float* outp = (float*)d_out;

  // -------- workspace carve-up (deterministic) --------
  char* wp = (char*)d_ws;
  auto alloc = [&](size_t bytes) { char* p = wp; wp += (bytes + 255) & ~(size_t)255; return p; };
  h16* pwE1 = (h16*)alloc((size_t)512 * 512 * 3 * 2);
  h16* pwE2 = (h16*)alloc((size_t)512 * 512 * 3 * 2);
  h16* pwD1 = (h16*)alloc((size_t)512 * 512 * 3 * 2);
  h16* pwD2 = (h16*)alloc((size_t)512 * 512 * 3 * 2);
  h16* pwMI = (h16*)alloc((size_t)128 * 512 * 2);
  h16* pwMO = (h16*)alloc((size_t)1024 * 128 * 2);
  h16* pwB1 = (h16*)alloc((size_t)NBLK * 32 * 128 * 2);
  h16* pwBC = (h16*)alloc((size_t)NBLK * 128 * 32 * 2);
  h16* pwW3 = (h16*)alloc((size_t)16 * 512 * 2);
  float* CB = (float*)alloc((size_t)NBLK * 128 * 4);
  h16* ACT0 = (h16*)alloc((size_t)8 * T0 * 512 * 2 + (size_t)8 * T1 * 512 * 2);
  h16* ACT1 = ACT0 + (size_t)8 * T0 * 512;
  h16* Y0   = ACT0;                         // reuse ACT0+ACT1 after encoder
  h16* Ff   = (h16*)alloc((size_t)8 * T2 * 512 * 2);
  float* O32 = (float*)alloc((size_t)8 * T2 * 128 * 4);
  h16* O16 = (h16*)alloc((size_t)8 * T2 * 128 * 2);
  h16* H1  = (h16*)alloc((size_t)8 * T2 * 32 * 2);
  h16* H2  = (h16*)alloc((size_t)8 * T2 * 32 * 2);
  size_t m16b = (size_t)8 * T2 * 1024 * 2, y2b = (size_t)16 * T0 * 512 * 2;
  h16* M16 = (h16*)alloc(m16b > y2b ? m16b : y2b);
  h16* Y2  = M16;                           // reuse after mask applied
  h16* Y1  = (h16*)alloc((size_t)16 * T1 * 512 * 2);

  // -------- weight packing --------
  launch_pack(stream, enc_w1, nullptr, pwE1, 512, 512, 3, 1536, 3, 1, 0);
  launch_pack(stream, enc_w2, nullptr, pwE2, 512, 512, 3, 1536, 3, 1, 0);
  launch_pack(stream, dec_w1, nullptr, pwD1, 512, 512, 3, 3, 1536, -1, 2);
  launch_pack(stream, dec_w2, nullptr, pwD2, 512, 512, 3, 3, 1536, -1, 2);
  launch_pack(stream, mg_in_w, nullptr, pwMI, 128, 512, 1, 512, 1, 0, 0);
  launch_pack(stream, mg_out_w, nullptr, pwMO, 1024, 128, 1, 128, 1, 0, 0);
  for (int i = 0; i < NBLK; ++i) {
    launch_pack(stream, blk_w1 + (size_t)i * 32 * 128, nullptr,
                pwB1 + (size_t)i * 32 * 128, 32, 128, 1, 128, 1, 0, 0);
    const float* s0 = (i < NBLK - 1) ? (blk_rw + (size_t)i * 128 * 32)
                                     : (blk_sw + (size_t)i * 128 * 32);
    const float* s1 = (i < NBLK - 1) ? (blk_sw + (size_t)i * 128 * 32) : nullptr;
    launch_pack(stream, s0, s1, pwBC + (size_t)i * 128 * 32, 128, 32, 1, 32, 1, 0, 0);
  }
  pack_w3_kernel<<<(16 * 512 + 255) / 256, 256, 0, stream>>>(dec_w3, pwW3);
  comb_bias_kernel<<<(NBLK * 128 + 255) / 256, 256, 0, stream>>>(blk_rb, blk_sb, CB);

  // -------- encoder --------
  {
    size_t tot = (size_t)8 * T0 * 512;
    enc0_kernel<<<(unsigned)((tot + 255) / 256), 256, 0, stream>>>(x_real, x_imag, enc_w0, ACT0);
  }
  launch_gemm(stream, ACT0, pwE1, nullptr, nullptr, nullptr, ACT1, nullptr, enc_a + 0,
              8, 512, 512, T0, T1, 3, 0, 0, 1);
  launch_gemm(stream, ACT1, pwE2, nullptr, nullptr, nullptr, Ff, nullptr, enc_a + 1,
              8, 512, 512, T1, T2, 3, 0, 0, 1);

  // -------- mask generator --------
  launch_gemm(stream, Ff, pwMI, mg_in_b, nullptr, O32, O16, nullptr, nullptr,
              8, 512, 128, T2, T2, 1, 0, 0, 0);
  for (int i = 0; i < NBLK; ++i) {
    int dil = 1 << (i & 7);
    launch_gemm(stream, O16, pwB1 + (size_t)i * 32 * 128, blk_b1 + i * 32, nullptr,
                nullptr, H1, nullptr, blk_a1 + i, 8, 128, 32, T2, T2, 1, 0, 0, 1);
    {
      size_t tot = (size_t)8 * T2 * 32;
      depthwise_kernel<<<(unsigned)((tot + 255) / 256), 256, 0, stream>>>(
          H1, blk_dw + (size_t)i * 96, blk_db + i * 32, blk_a2 + i, H2, T2, dil);
    }
    bool last = (i == NBLK - 1);
    launch_gemm(stream, H2, pwBC + (size_t)i * 128 * 32, CB + i * 128, O32,
                O32, O16, nullptr, last ? mg_out_a : nullptr,
                8, 32, 128, T2, T2, 1, 0, 0, last ? 1 : 0);
  }
  launch_gemm(stream, O16, pwMO, mg_out_b, nullptr, nullptr, M16, nullptr, nullptr,
              8, 128, 1024, T2, T2, 1, 0, 0, 2);
  {
    size_t tot = (size_t)16 * T2 * 512;
    mask_kernel<<<(unsigned)((tot + 255) / 256), 256, 0, stream>>>(M16, Ff, Y0);
  }

  // -------- decoder --------
  launch_gemm(stream, Y0, pwD1, nullptr, nullptr, nullptr, Y1, nullptr, dec_a + 0,
              16, 512, 512, T2, T1, 3, 2, 0, 1);
  launch_gemm(stream, Y1, pwD2, nullptr, nullptr, nullptr, Y2, nullptr, dec_a + 1,
              16, 512, 512, T1, T0, 3, 2, 0, 1);
  dec_final_kernel<<<16 * T_RAW / 8, 256, 0, stream>>>(Y2, pwW3, outp);
}
